// dynamic_graph_constructor_79817672229087
// MI455X (gfx1250) — matrix operand order repaired and checked by reading, not compiled, not run on hardware
//
#include <hip/hip_runtime.h>
#include <hip/hip_bf16.h>
#include <math.h>

// ---------------- problem constants ----------------
#define N_NODES 300
#define NP      320          // padded node count (multiple of 32 for WMMA K)
#define D_EMB   32
#define B_SZ    16
#define C_CH    32
#define L_SEQ   12
#define CIN     96
#define BL      (B_SZ * L_SEQ)        // 192
#define RS      (B_SZ * C_CH * L_SEQ) // 6144 rows of the spatial GEMM
#define ALPHA_F 3.0f

typedef _Float16 half_t;
typedef __attribute__((ext_vector_type(16))) _Float16 v16h;
typedef __attribute__((ext_vector_type(8)))  _Float16 v8h;
typedef __attribute__((ext_vector_type(8)))  float    v8f;

// Load a 16-f16 WMMA fragment slice from two 16-byte-aligned chunks.
__device__ __forceinline__ v16h ld_frag(const half_t* p0, const half_t* p1) {
    union { v16h v; v8h h[2]; } u;
    u.h[0] = *(const v8h*)p0;
    u.h[1] = *(const v8h*)p1;
    return u.v;
}

__device__ __forceinline__ v8f wmma_f16(v16h a, v16h b, v8f c) {
    // D = A(16x32 f16) * B(32x16 f16) + C(16x16 f32)
    return __builtin_amdgcn_wmma_f32_16x16x32_f16(false, a, false, b,
                                                  (short)0, c, false, false);
}

// ---------------- kernels ----------------

__global__ void k_zero(uint4* p, size_t n16) {
    size_t i = (size_t)blockIdx.x * blockDim.x + threadIdx.x;
    size_t stride = (size_t)gridDim.x * blockDim.x;
    uint4 z; z.x = z.y = z.z = z.w = 0u;
    for (; i < n16; i += stride) p[i] = z;
}

// nodevec: nv = tanh(3*(emb @ W^T + b)); 600 blocks (300 nv1 + 300 nv2), 32 thr
__global__ void k_nodevec(const float* emb1, const float* emb2,
                          const float* w1, const float* b1,
                          const float* w2, const float* b2,
                          float* nv1f, float* nv2f, half_t* nv1h, half_t* nv2h) {
    int g = blockIdx.x, d = threadIdx.x;
    const float *emb, *w, *bb; float* of; half_t* oh; int row;
    if (g < N_NODES) { emb = emb1; w = w1; bb = b1; of = nv1f; oh = nv1h; row = g; }
    else             { emb = emb2; w = w2; bb = b2; of = nv2f; oh = nv2h; row = g - N_NODES; }
    float s = bb[d];
    for (int k = 0; k < D_EMB; ++k) s += emb[row * D_EMB + k] * w[d * D_EMB + k];
    float v = tanhf(ALPHA_F * s);
    of[row * D_EMB + d] = v;
    oh[row * D_EMB + d] = (half_t)v;
}

__global__ void k_sqnorm(const float* nv1f, const float* nv2f,
                         float* n1sq, float* n2sq) {
    int g = blockIdx.x, d = threadIdx.x;
    const float* nv = (g < N_NODES) ? nv1f : nv2f;
    float* out = (g < N_NODES) ? n1sq : n2sq;
    int row = (g < N_NODES) ? g : g - N_NODES;
    float x = nv[row * D_EMB + d];
    float s = x * x;
    for (int off = 16; off >= 1; off >>= 1) s += __shfl_xor(s, off, 32);
    if (d == 0) out[row] = s;
}

__global__ void k_f2h(const float* a, const float* b, half_t* ah, half_t* bh, int n) {
    int i = blockIdx.x * blockDim.x + threadIdx.x;
    if (i < n) { ah[i] = (half_t)a[i]; bh[i] = (half_t)b[i]; }
}

// Gaussian kernel matrix via WMMA: G[i,j] = exp(-(|a|^2+|b|^2-2 a.b)/18)
#define GT 19   // ceil(300/16)
__global__ void k_gauss(const half_t* nv1h, const half_t* nv2h,
                        const float* n1sq, const float* n2sq, float* G) {
    int wave = blockIdx.x * (blockDim.x >> 5) + (threadIdx.x >> 5);
    int lane = threadIdx.x & 31;
    if (wave >= GT * GT) return;
    int ti = wave / GT, tj = wave % GT;
    int m = lane & 15, hl = lane >> 4;
    const half_t* ar = nv1h + (ti * 16 + m) * D_EMB;           // A: rows, K-contig
    v16h a = ld_frag(ar + 16 * hl, ar + 16 * hl + 8);
    const half_t* br = nv2h + (tj * 16 + m) * D_EMB + 16 * hl; // B[k,n]=nv2[col n,k]
    v16h b = ld_frag(br, br + 8);
    v8f c = {};
    c = wmma_f16(a, b, c);
    int j = tj * 16 + m;
    for (int r = 0; r < 8; ++r) {
        int i = ti * 16 + r + 8 * hl;
        if (i < N_NODES && j < N_NODES) {
            float d2 = n1sq[i] + n2sq[j] - 2.0f * c[r];
            G[i * NP + j] = __expf(-d2 * (1.0f / (2.0f * ALPHA_F * ALPHA_F)));
        }
    }
}

// rs[v] = sum_w G[v,w] (g<300), cs[v] = sum_w G[w,v] (g>=300)
__global__ void k_rowcolsum(const float* G, float* rs, float* cs) {
    int g = blockIdx.x, lane = threadIdx.x;
    float s = 0.f;
    if (g < N_NODES) { for (int j = lane; j < N_NODES; j += 32) s += G[g * NP + j]; }
    else { int v = g - N_NODES; for (int j = lane; j < N_NODES; j += 32) s += G[j * NP + v]; }
    for (int off = 16; off >= 1; off >>= 1) s += __shfl_xor(s, off, 32);
    if (lane == 0) { if (g < N_NODES) rs[g] = s; else cs[g - N_NODES] = s; }
}

// A1hT[w][v] = G[v,w]/rs[v]; A2hT[w][v] = G[w,v]/cs[v]  (K-contiguous B operands)
__global__ void k_buildA(const float* G, const float* rs, const float* cs,
                         half_t* A1hT, half_t* A2hT) {
    int i = blockIdx.x * blockDim.x + threadIdx.x;
    if (i >= N_NODES * N_NODES) return;
    int v = i / N_NODES, w = i % N_NODES;
    float rv = rs[v], cv = cs[v];
    float r1 = (rv > 0.f) ? 1.f / rv : 0.f;
    float r2 = (cv > 0.f) ? 1.f / cv : 0.f;
    A1hT[w * NP + v] = (half_t)(G[v * NP + w] * r1);
    A2hT[w * NP + v] = (half_t)(G[w * NP + v] * r2);
}

// x [B,C,N,L] -> xt[(b,c,l)][v] f16  and xc[(b,l)][v][c] f16 (zero-padded v)
__global__ void k_xtrans(const float* x, half_t* xt, half_t* xc) {
    int idx = blockIdx.x * blockDim.x + threadIdx.x;
    if (idx >= RS * NP) return;
    int v = idx % NP, r = idx / NP;               // r = (b*C + c)*L + l
    int l = r % L_SEQ, t = r / L_SEQ;
    int c = t % C_CH, b = t / C_CH;
    float val = (v < N_NODES) ? x[((b * C_CH + c) * N_NODES + v) * L_SEQ + l] : 0.f;
    half_t h = (half_t)val;
    xt[r * NP + v] = h;
    xc[((b * L_SEQ + l) * NP + v) * C_CH + c] = h;
}

// spatial hop: out[row,w] = sum_v S[row,v] * A[v,w]   (A given transposed)
// 4 column tiles per wave: A fragment loaded once per K-step, reused 4x.
#define SP_CT 4
__global__ void k_spmm(const half_t* S, const half_t* AhT,
                       half_t* outT, half_t* outC) {
    int wave = blockIdx.x * (blockDim.x >> 5) + (threadIdx.x >> 5);
    int lane = threadIdx.x & 31;
    const int MT = RS / 16;                 // 384 row tiles
    const int NG = (NP / 16) / SP_CT;       // 5 groups of 4 column tiles
    if (wave >= MT * NG) return;
    int tm = wave / NG, tg = wave % NG;
    int m = lane & 15, hl = lane >> 4;
    const half_t* arow  = S + (tm * 16 + m) * NP;
    const half_t* brow0 = AhT + (tg * SP_CT * 16 + m) * NP + 16 * hl;
    v8f acc0 = {}, acc1 = {}, acc2 = {}, acc3 = {};
    for (int k0 = 0; k0 < NP; k0 += 32) {
        v16h a  = ld_frag(arow + k0 + 16 * hl, arow + k0 + 16 * hl + 8);
        v16h b0 = ld_frag(brow0 + k0,             brow0 + k0 + 8);
        v16h b1 = ld_frag(brow0 + 16 * NP + k0,   brow0 + 16 * NP + k0 + 8);
        v16h b2 = ld_frag(brow0 + 32 * NP + k0,   brow0 + 32 * NP + k0 + 8);
        v16h b3 = ld_frag(brow0 + 48 * NP + k0,   brow0 + 48 * NP + k0 + 8);
        acc0 = wmma_f16(a, b0, acc0);
        acc1 = wmma_f16(a, b1, acc1);
        acc2 = wmma_f16(a, b2, acc2);
        acc3 = wmma_f16(a, b3, acc3);
    }
    int wbase = tg * SP_CT * 16 + m;
    for (int r = 0; r < 8; ++r) {
        int rg = tm * 16 + r + 8 * hl;
        float vals[SP_CT] = { acc0[r], acc1[r], acc2[r], acc3[r] };
        if (outT) {
            for (int j = 0; j < SP_CT; ++j)
                outT[rg * NP + wbase + j * 16] = (half_t)vals[j];
        }
        if (outC) {
            int l = rg % L_SEQ, t = rg / L_SEQ;
            int c = t % C_CH, b = t / C_CH;
            size_t cbase = (size_t)(b * L_SEQ + l) * NP * C_CH + c;
            for (int j = 0; j < SP_CT; ++j)
                outC[cbase + (size_t)(wbase + j * 16) * C_CH] = (half_t)vals[j];
        }
    }
}

// projection fused with bias + tanh(3*nv*filt) -> nvb f16 [(bl)][node][c]
// Both o-tiles per wave: hc A-fragment loaded once per part, reused 2x.
__global__ void k_filt_nvb(const half_t* xc, const half_t* h1c, const half_t* h2c,
                           const half_t* Wh, const float* bias,
                           const float* nvf, half_t* nvb) {
    int wave = blockIdx.x * (blockDim.x >> 5) + (threadIdx.x >> 5);
    int lane = threadIdx.x & 31;
    const int NTILE = NP / 16;              // 20
    if (wave >= BL * NTILE) return;
    int nt = wave % NTILE, bl = wave / NTILE;
    int m = lane & 15, hl = lane >> 4;
    const half_t* srcs[3] = { xc, h1c, h2c };
    v8f acc0 = {}, acc1 = {};
    for (int part = 0; part < 3; ++part) {
        const half_t* arow = srcs[part] + ((bl * NP) + nt * 16 + m) * C_CH;
        v16h a = ld_frag(arow + 16 * hl, arow + 16 * hl + 8);
        const half_t* br0 = Wh + m * CIN + part * 32 + 16 * hl;
        const half_t* br1 = Wh + (16 + m) * CIN + part * 32 + 16 * hl;
        acc0 = wmma_f16(a, ld_frag(br0, br0 + 8), acc0);
        acc1 = wmma_f16(a, ld_frag(br1, br1 + 8), acc1);
    }
    int o0 = m, o1 = 16 + m;
    float bo0 = bias[o0], bo1 = bias[o1];
    for (int r = 0; r < 8; ++r) {
        int node = nt * 16 + r + 8 * hl;
        size_t base = ((size_t)(bl * NP) + node) * C_CH;
        float v0 = acc0[r] + bo0;
        float v1 = acc1[r] + bo1;
        nvb[base + o0] = (half_t)tanhf(ALPHA_F * nvf[node * D_EMB + o0] * v0);
        nvb[base + o1] = (half_t)tanhf(ALPHA_F * nvf[node * D_EMB + o1] * v1);
    }
}

// batched antisymmetric GEMM + relu(tanh): grid is exactly BL*190/8 blocks
#define AT 19
__global__ void k_adj(const half_t* nv1b, const half_t* nv2b, float* out1) {
    __shared__ float lds[8 * 256];
    int wslot = threadIdx.x >> 5;
    int wave = blockIdx.x * (blockDim.x >> 5) + wslot;
    int lane = threadIdx.x & 31;
    const int NPAIR = AT * (AT + 1) / 2;           // 190
    int p = wave % NPAIR, bl = wave / NPAIR;
    int tn = 0, rem = p;
    while (rem >= (AT - tn)) { rem -= (AT - tn); ++tn; }
    int tm = tn + rem;
    int m = lane & 15, hl = lane >> 4;
    const half_t* b1p = nv1b + (size_t)bl * NP * C_CH;
    const half_t* b2p = nv2b + (size_t)bl * NP * C_CH;
    // P = nv1b[tn tile] x nv2b[tm tile]^T
    const half_t* a1 = b1p + (tn * 16 + m) * C_CH;
    const half_t* bb1 = b2p + (tm * 16 + m) * C_CH + 16 * hl;
    v8f P = {};
    P = wmma_f16(ld_frag(a1 + 16 * hl, a1 + 16 * hl + 8), ld_frag(bb1, bb1 + 8), P);
    // Q = nv1b[tm tile] x nv2b[tn tile]^T
    const half_t* a2 = b1p + (tm * 16 + m) * C_CH;
    const half_t* bb2 = b2p + (tn * 16 + m) * C_CH + 16 * hl;
    v8f Q = {};
    Q = wmma_f16(ld_frag(a2 + 16 * hl, a2 + 16 * hl + 8), ld_frag(bb2, bb2 + 8), Q);
    // transpose Q through per-wave LDS tile
    float* t = lds + wslot * 256;
    for (int r = 0; r < 8; ++r) t[(r + 8 * hl) * 16 + m] = Q[r];
    __syncthreads();
    float* outbl = out1 + (size_t)bl * N_NODES * N_NODES;
    for (int r = 0; r < 8; ++r) {
        float qT = t[m * 16 + (r + 8 * hl)];
        float th = tanhf(ALPHA_F * (P[r] - qT));   // a = P - Q^T
        int i = tn * 16 + r + 8 * hl, j = tm * 16 + m;
        if (i < N_NODES && j < N_NODES) outbl[i * N_NODES + j] = fmaxf(th, 0.f);
        int i2 = tm * 16 + m, j2 = tn * 16 + r + 8 * hl;   // a(tm,tn) = -a(tn,tm)^T
        if (i2 < N_NODES && j2 < N_NODES) outbl[i2 * N_NODES + j2] = fmaxf(-th, 0.f);
    }
}

// exact top-20 per row (ties -> lowest index, like jax.lax.top_k), in place
__global__ void k_topk(float* out1) {
    int wave = blockIdx.x * (blockDim.x >> 5) + (threadIdx.x >> 5);
    int lane = threadIdx.x & 31;
    if (wave >= BL * N_NODES) return;
    float* row = out1 + (size_t)wave * N_NODES;
    float v[10];
#pragma unroll
    for (int s = 0; s < 10; ++s) {
        int j = lane + 32 * s;
        v[s] = (j < N_NODES) ? row[j] : -1.f;
    }
    unsigned kept = 0;
    for (int it = 0; it < 20; ++it) {
        float bv = -2.f; int bi = 1 << 30;
#pragma unroll
        for (int s = 0; s < 10; ++s) {
            int j = lane + 32 * s;
            if (j < N_NODES && !((kept >> s) & 1) && v[s] > bv) { bv = v[s]; bi = j; }
        }
        for (int off = 16; off >= 1; off >>= 1) {
            float ov = __shfl_xor(bv, off, 32);
            int   oi = __shfl_xor(bi, off, 32);
            if (ov > bv || (ov == bv && oi < bi)) { bv = ov; bi = oi; }
        }
        if (lane == (bi & 31)) kept |= 1u << (bi >> 5);
    }
#pragma unroll
    for (int s = 0; s < 10; ++s) {
        int j = lane + 32 * s;
        if (j < N_NODES) row[j] = ((kept >> s) & 1) ? v[s] : 0.f;
    }
}

// prep1: row = (row + I) / rowsum, in place
__global__ void k_prep1(float* out1) {
    int wave = blockIdx.x * (blockDim.x >> 5) + (threadIdx.x >> 5);
    int lane = threadIdx.x & 31;
    if (wave >= BL * N_NODES) return;
    int n = wave % N_NODES;
    float* row = out1 + (size_t)wave * N_NODES;
    float s = 0.f;
    for (int j = lane; j < N_NODES; j += 32) s += row[j];
    for (int off = 16; off >= 1; off >>= 1) s += __shfl_xor(s, off, 32);
    float inv = 1.f / (s + 1.f);
    for (int j = lane; j < N_NODES; j += 32)
        row[j] = (row[j] + ((j == n) ? 1.f : 0.f)) * inv;
}

// prep2: out2[n,m] = (out1[m,n] + I) / (colsum_n(out1) + 1)
__global__ void k_prep2(const float* out1, float* out2) {
    int wave = blockIdx.x * (blockDim.x >> 5) + (threadIdx.x >> 5);
    int lane = threadIdx.x & 31;
    if (wave >= BL * N_NODES) return;
    int bl = wave / N_NODES, n = wave % N_NODES;
    const float* src = out1 + (size_t)bl * N_NODES * N_NODES;
    float s = 0.f;
    for (int mm = lane; mm < N_NODES; mm += 32) s += src[mm * N_NODES + n];
    for (int off = 16; off >= 1; off >>= 1) s += __shfl_xor(s, off, 32);
    float inv = 1.f / (s + 1.f);
    float* dst = out2 + (size_t)bl * N_NODES * N_NODES + (size_t)n * N_NODES;
    for (int mm = lane; mm < N_NODES; mm += 32)
        dst[mm] = (src[mm * N_NODES + n] + ((mm == n) ? 1.f : 0.f)) * inv;
}

// ---------------- host launch ----------------
extern "C" void kernel_launch(void* const* d_in, const int* in_sizes, int n_in,
                              void* d_out, int out_size, void* d_ws, size_t ws_size,
                              hipStream_t stream) {
    const float* x      = (const float*)d_in[0];
    // d_in[1] = idx (identity arange) — unused
    const float* emb1   = (const float*)d_in[2];
    const float* emb2   = (const float*)d_in[3];
    const float* lin1_w = (const float*)d_in[4];
    const float* lin1_b = (const float*)d_in[5];
    const float* lin2_w = (const float*)d_in[6];
    const float* lin2_b = (const float*)d_in[7];
    const float* f1_w   = (const float*)d_in[8];
    const float* f1_b   = (const float*)d_in[9];
    const float* f2_w   = (const float*)d_in[10];
    const float* f2_b   = (const float*)d_in[11];

    float* out1 = (float*)d_out;
    float* out2 = out1 + (size_t)BL * N_NODES * N_NODES;

    // workspace carve-out (all regions 256B aligned)
    char* ws = (char*)d_ws;
    size_t off = 0;
    auto take = [&](size_t bytes) -> char* {
        char* p = ws + off;
        off += (bytes + 255) & ~(size_t)255;
        return p;
    };
    float*  nv1f = (float*) take(NP * D_EMB * 4);
    float*  nv2f = (float*) take(NP * D_EMB * 4);
    half_t* nv1h = (half_t*)take(NP * D_EMB * 2);
    half_t* nv2h = (half_t*)take(NP * D_EMB * 2);
    float*  n1sq = (float*) take(NP * 4);
    float*  n2sq = (float*) take(NP * 4);
    float*  G    = (float*) take((size_t)NP * NP * 4);
    float*  rsum = (float*) take(NP * 4);
    float*  csum = (float*) take(NP * 4);
    half_t* A1hT = (half_t*)take((size_t)NP * NP * 2);
    half_t* A2hT = (half_t*)take((size_t)NP * NP * 2);
    half_t* W1h  = (half_t*)take(C_CH * CIN * 2);
    half_t* W2h  = (half_t*)take(C_CH * CIN * 2);
    half_t* xt   = (half_t*)take((size_t)RS * NP * 2);
    half_t* x1t  = (half_t*)take((size_t)RS * NP * 2);
    half_t* xc   = (half_t*)take((size_t)BL * NP * C_CH * 2);
    half_t* x1c  = (half_t*)take((size_t)BL * NP * C_CH * 2);
    half_t* x2c  = (half_t*)take((size_t)BL * NP * C_CH * 2);
    half_t* x1c2 = (half_t*)take((size_t)BL * NP * C_CH * 2);
    half_t* x2c2 = (half_t*)take((size_t)BL * NP * C_CH * 2);
    half_t* nv1b = (half_t*)take((size_t)BL * NP * C_CH * 2);
    half_t* nv2b = (half_t*)take((size_t)BL * NP * C_CH * 2);
    size_t used = off;

    // 0) zero the used workspace (establishes all zero padding)
    k_zero<<<2048, 256, 0, stream>>>((uint4*)d_ws, used / 16);

    // 1) node vectors + norms, f16 weight copies
    k_nodevec<<<2 * N_NODES, 32, 0, stream>>>(emb1, emb2, lin1_w, lin1_b,
                                              lin2_w, lin2_b, nv1f, nv2f, nv1h, nv2h);
    k_sqnorm<<<2 * N_NODES, 32, 0, stream>>>(nv1f, nv2f, n1sq, n2sq);
    k_f2h<<<(C_CH * CIN + 255) / 256, 256, 0, stream>>>(f1_w, f2_w, W1h, W2h, C_CH * CIN);

    // 2) Gaussian adjacency + normalizations (WMMA)
    k_gauss<<<(GT * GT + 7) / 8, 256, 0, stream>>>(nv1h, nv2h, n1sq, n2sq, G);
    k_rowcolsum<<<2 * N_NODES, 32, 0, stream>>>(G, rsum, csum);
    k_buildA<<<(N_NODES * N_NODES + 255) / 256, 256, 0, stream>>>(G, rsum, csum, A1hT, A2hT);

    // 3) input transposes
    k_xtrans<<<(RS * NP + 255) / 256, 256, 0, stream>>>(x, xt, xc);

    // 4) GCN spatial hops (WMMA, 4 col-tiles/wave), filter1 then filter2
    const int SPBLK = (RS / 16) * ((NP / 16) / SP_CT) / 8;   // 240
    k_spmm<<<SPBLK, 256, 0, stream>>>(xt,  A1hT, x1t, x1c);
    k_spmm<<<SPBLK, 256, 0, stream>>>(x1t, A1hT, (half_t*)nullptr, x2c);
    k_spmm<<<SPBLK, 256, 0, stream>>>(xt,  A2hT, x1t, x1c2);
    k_spmm<<<SPBLK, 256, 0, stream>>>(x1t, A2hT, (half_t*)nullptr, x2c2);

    // 5) channel projection fused with bias + tanh(3*nv*filt) (WMMA, 2 o-tiles/wave)
    const int PRBLK = (BL * (NP / 16)) / 8;                  // 480
    k_filt_nvb<<<PRBLK, 256, 0, stream>>>(xc, x1c,  x2c,  W1h, f1_b, nv1f, nv1b);
    k_filt_nvb<<<PRBLK, 256, 0, stream>>>(xc, x1c2, x2c2, W2h, f2_b, nv2f, nv2b);

    // 6) batched antisymmetric GEMM + relu(tanh) (WMMA + LDS transpose)
    const int ADJBLK = (BL * (AT * (AT + 1) / 2)) / 8;       // 4560, exact
    k_adj<<<ADJBLK, 256, 0, stream>>>(nv1b, nv2b, out1);

    // 7) top-20 mask, then both normalizations
    const int ROWBLK = (BL * N_NODES + 7) / 8;               // 7200
    k_topk <<<ROWBLK, 256, 0, stream>>>(out1);
    k_prep1<<<ROWBLK, 256, 0, stream>>>(out1);
    k_prep2<<<ROWBLK, 256, 0, stream>>>(out1, out2);
}
